// TextRNN_44281112822509
// MI455X (gfx1250) — compile-verified, matrix operand order checked by reading, not run on hardware
//
#include <hip/hip_runtime.h>
#include <hip/hip_bf16.h>
#include <math.h>

#define N_CLASS 32000
#define EMB     256
#define HID     512
#define BATCH   128
#define SEQ     512
#define NWG     32      // workgroups in persistent RNN kernel (each owns 16 hidden columns)
#define PAD     16      // bf16 elements of padding per LDS row (32B) to stagger banks

typedef __attribute__((ext_vector_type(16))) __bf16 v16bf;
typedef __attribute__((ext_vector_type(8)))  float  v8f;
typedef __attribute__((ext_vector_type(2)))  float  v2f;

// ---- explicit global-address-space accessors: force global_load_b128 /
// ---- global_store_b16 instead of flat_* (flat ties up the LDS path and
// ---- couples the WMMA stream to DScnt waits).
__device__ __forceinline__ v16bf ldg_v16bf(const __bf16* p) {
  typedef v16bf __attribute__((address_space(1))) gvec;
  return *(const gvec*)p;
}
__device__ __forceinline__ v2f ldg_v2f(const float* p) {
  typedef v2f __attribute__((address_space(1))) gvec;
  return *(const gvec*)p;
}
__device__ __forceinline__ void stg_bf16(__bf16* p, __bf16 v) {
  typedef __bf16 __attribute__((address_space(1))) gb;
  *(gb*)p = v;
}
__device__ __forceinline__ void stg_f32(float* p, float v) {
  typedef float __attribute__((address_space(1))) gf;
  *(gf*)p = v;
}

__device__ __forceinline__ v8f wmma_bf16(v16bf a, v16bf b, v8f c) {
  // D = A(16x32 bf16) * B(32x16 bf16) + C(16x16 f32)
  return __builtin_amdgcn_wmma_f32_16x16x32_bf16(false, a, false, b, (short)0, c, false, false);
}

// ---------------------------------------------------------------------------
// Embedding gather: emb_bf16[t][b][e] = (bf16) C[X[b][t]][e]   (time-major)
// ---------------------------------------------------------------------------
__global__ void gather_emb_kernel(const int* __restrict__ X,
                                  const float* __restrict__ C,
                                  __bf16* __restrict__ emb) {
  const int b = blockIdx.x >> 9;         // / SEQ
  const int t = blockIdx.x & (SEQ - 1);  // % SEQ
  const int e = threadIdx.x;             // 256 threads = EMB
  const int idx = X[(size_t)b * SEQ + t];
  stg_bf16(&emb[((size_t)t * BATCH + b) * EMB + e],
           (__bf16)C[(size_t)idx * EMB + e]);
}

// ---------------------------------------------------------------------------
// Zero the ping-pong hidden-state buffers (contiguous) + barrier counters.
// Harness poisons ws with 0xAA and never re-poisons, so do this every call.
// ---------------------------------------------------------------------------
__global__ void init_ws_kernel(unsigned* __restrict__ ctrs,
                               unsigned* __restrict__ hzero, int nwords) {
  const int i = blockIdx.x * blockDim.x + threadIdx.x;
  if (i < nwords) hzero[i] = 0u;   // bf16 +0.0 == 0x0000
  if (i < 2) ctrs[i] = 0u;         // [0]=arrive count, [1]=generation
}

// ---------------------------------------------------------------------------
// Software grid barrier (persistent kernel, NWG co-resident workgroups)
// ---------------------------------------------------------------------------
__device__ __forceinline__ void gridBarrier(unsigned* cnt, unsigned* gen) {
  __syncthreads();
  if (threadIdx.x == 0) {
    __threadfence();  // make our h-state stores device-visible
    unsigned g = __hip_atomic_load(gen, __ATOMIC_RELAXED, __HIP_MEMORY_SCOPE_AGENT);
    unsigned a = __hip_atomic_fetch_add(cnt, 1u, __ATOMIC_ACQ_REL, __HIP_MEMORY_SCOPE_AGENT);
    if (a == NWG - 1) {
      __hip_atomic_store(cnt, 0u, __ATOMIC_RELAXED, __HIP_MEMORY_SCOPE_AGENT);
      __hip_atomic_fetch_add(gen, 1u, __ATOMIC_RELEASE, __HIP_MEMORY_SCOPE_AGENT);
    } else {
      while (__hip_atomic_load(gen, __ATOMIC_ACQUIRE, __HIP_MEMORY_SCOPE_AGENT) == g)
        __builtin_amdgcn_s_sleep(2);
    }
    __threadfence();
  }
  __syncthreads();
}

// ---------------------------------------------------------------------------
// Persistent RNN scan. Grid = NWG(32) x 256 threads (8 wave32).
// WG owns hidden columns [16*blockIdx.x, +16); its weight rows live in LDS.
// Wave w owns batch rows [16w, +16). One 16x16 WMMA tile per wave per layer.
// One grid barrier per timestep (h1_t and h0_{t+1} only need h0_t + older).
// ---------------------------------------------------------------------------
__global__ void __launch_bounds__(256) rnn_persistent_kernel(
    const float* __restrict__ W_ax, const float* __restrict__ W_aa,
    const float* __restrict__ W_aal, const float* __restrict__ b_a,
    const __bf16* __restrict__ emb,
    __bf16* __restrict__ h0_0, __bf16* __restrict__ h0_1,
    __bf16* __restrict__ h1_0, __bf16* __restrict__ h1_1,
    float* __restrict__ h1f, unsigned* __restrict__ ctrs) {
  __shared__ __align__(64) __bf16 sWax[16][EMB + PAD];  //  8.5 KB
  __shared__ __align__(64) __bf16 sWaa[16][HID + PAD];  // 16.5 KB
  __shared__ __align__(64) __bf16 sWal[16][HID + PAD];  // 16.5 KB
  __shared__ float sBias[16];

  const int tid  = threadIdx.x;
  const int lane = tid & 31;
  const int wave = tid >> 5;          // m-tile 0..7
  const int nb   = blockIdx.x * 16;   // first hidden column owned by this WG

  // Stage this WG's weight slice into LDS as bf16 (rows nb..nb+15).
  for (int i = tid; i < 16 * EMB; i += 256) {
    const int r = i >> 8, c = i & (EMB - 1);
    sWax[r][c] = (__bf16)W_ax[(size_t)(nb + r) * EMB + c];
  }
  for (int i = tid; i < 16 * HID; i += 256) {
    const int r = i >> 9, c = i & (HID - 1);
    sWaa[r][c] = (__bf16)W_aa[(size_t)(nb + r) * HID + c];
    sWal[r][c] = (__bf16)W_aal[(size_t)(nb + r) * HID + c];
  }
  if (tid < 16) sBias[tid] = b_a[nb + tid];
  __syncthreads();

  const int nl   = lane & 15;               // B column / output column (local)
  const int ks   = (lane >> 4) * 16;        // bf16 A/B K-select: half h holds k 16h..16h+15 (contiguous)
  const int mrow = wave * 16 + nl;          // A-matrix row (batch index)
  const int orow = wave * 16 + ((lane >> 4) << 3);  // C/D row base (acc[0])
  const float bias = sBias[nl];

  for (int t = 0; t < SEQ; ++t) {
    const int wr = t & 1;
    // ping-pong via ternaries on kernel args (keeps address-space inference)
    const __bf16* h0r = wr ? h0_0 : h0_1;   // h0_{t-1}
    __bf16*       h0w = wr ? h0_1 : h0_0;   // h0_t
    const __bf16* h1r = wr ? h1_0 : h1_1;   // h1_{t-1}
    __bf16*       h1w = wr ? h1_1 : h1_0;   // h1_t

    // ---- layer 0: h0_t = tanh(x_t @ Wax^T + h0_{t-1} @ Waa^T + b_a)
    {
      v8f a0 = {}, a1 = {};
      const __bf16* A = emb + ((size_t)t * BATCH + mrow) * EMB + ks;
#pragma unroll
      for (int k = 0; k < EMB; k += 64) {
        a0 = wmma_bf16(ldg_v16bf(A + k),
                       *(const v16bf*)(&sWax[nl][k + ks]), a0);
        a1 = wmma_bf16(ldg_v16bf(A + k + 32),
                       *(const v16bf*)(&sWax[nl][k + 32 + ks]), a1);
      }
      const __bf16* Ah = h0r + (size_t)mrow * HID + ks;
#pragma unroll
      for (int k = 0; k < HID; k += 64) {
        a0 = wmma_bf16(ldg_v16bf(Ah + k),
                       *(const v16bf*)(&sWaa[nl][k + ks]), a0);
        a1 = wmma_bf16(ldg_v16bf(Ah + k + 32),
                       *(const v16bf*)(&sWaa[nl][k + 32 + ks]), a1);
      }
      if (t + 1 < SEQ)  // warm caches for next step's A rows while we sync
        __builtin_prefetch(emb + ((size_t)(t + 1) * BATCH + mrow) * EMB + ks, 0, 3);
      const v8f acc = a0 + a1;
      __bf16* dst = h0w + nb + nl;
#pragma unroll
      for (int r = 0; r < 8; ++r)
        stg_bf16(&dst[(size_t)(orow + r) * HID], (__bf16)tanhf(acc[r] + bias));
    }

    gridBarrier(&ctrs[0], &ctrs[1]);  // h0_t now device-visible

    // ---- layer 1: h1_t = tanh(h0_t @ Waal^T + h1_{t-1} @ Waa^T + b_a)
    {
      v8f a0 = {}, a1 = {};
      const __bf16* A0 = h0w + (size_t)mrow * HID + ks;
#pragma unroll
      for (int k = 0; k < HID; k += 64) {
        a0 = wmma_bf16(ldg_v16bf(A0 + k),
                       *(const v16bf*)(&sWal[nl][k + ks]), a0);
        a1 = wmma_bf16(ldg_v16bf(A0 + k + 32),
                       *(const v16bf*)(&sWal[nl][k + 32 + ks]), a1);
      }
      const __bf16* A1 = h1r + (size_t)mrow * HID + ks;
#pragma unroll
      for (int k = 0; k < HID; k += 64) {
        a0 = wmma_bf16(ldg_v16bf(A1 + k),
                       *(const v16bf*)(&sWaa[nl][k + ks]), a0);
        a1 = wmma_bf16(ldg_v16bf(A1 + k + 32),
                       *(const v16bf*)(&sWaa[nl][k + 32 + ks]), a1);
      }
      const v8f acc = a0 + a1;
      __bf16* dst = h1w + nb + nl;
#pragma unroll
      for (int r = 0; r < 8; ++r) {
        const float v = tanhf(acc[r] + bias);
        stg_bf16(&dst[(size_t)(orow + r) * HID], (__bf16)v);
        if (t == SEQ - 1)  // fp32 copy for the full-precision projection
          stg_f32(&h1f[(size_t)(orow + r) * HID + nb + nl], v);
      }
    }
    // next step's h0 compute only reads h0_t (already fenced) + emb
  }
}

// ---------------------------------------------------------------------------
// Logits: out = h1_final @ W_out^T + b_out, fp32 WMMA 16x16x4.
// Grid = 2000 blocks (one 16-col N-tile each) x 8 waves (one 16-row M-tile).
// ---------------------------------------------------------------------------
__global__ void __launch_bounds__(256) proj_kernel(
    const float* __restrict__ h1f, const float* __restrict__ W_out,
    const float* __restrict__ b_out, float* __restrict__ out) {
  const int tid  = threadIdx.x;
  const int lane = tid & 31;
  const int wave = tid >> 5;            // m-tile 0..7
  const int nb   = blockIdx.x * 16;
  const int nl   = lane & 15;
  const int koff = (lane >> 4) * 2;     // f32 A/B K-select per ISA layout

  const float* A  = h1f  + (size_t)(wave * 16 + nl) * HID + koff;
  const float* Bp = W_out + (size_t)(nb + nl) * HID + koff;

  v8f a0 = {}, a1 = {};
#pragma unroll 8
  for (int k = 0; k < HID; k += 8) {
    a0 = __builtin_amdgcn_wmma_f32_16x16x4_f32(false, ldg_v2f(A + k),
                                               false, ldg_v2f(Bp + k),
                                               (short)0, a0, false, false);
    a1 = __builtin_amdgcn_wmma_f32_16x16x4_f32(false, ldg_v2f(A + k + 4),
                                               false, ldg_v2f(Bp + k + 4),
                                               (short)0, a1, false, false);
  }
  const v8f acc = a0 + a1;
  const float bias = b_out[nb + nl];
  const int orow = wave * 16 + ((lane >> 4) << 3);
#pragma unroll
  for (int r = 0; r < 8; ++r)
    stg_f32(&out[(size_t)(orow + r) * N_CLASS + nb + nl], acc[r] + bias);
}

// ---------------------------------------------------------------------------
extern "C" void kernel_launch(void* const* d_in, const int* in_sizes, int n_in,
                              void* d_out, int out_size, void* d_ws, size_t ws_size,
                              hipStream_t stream) {
  const int*   X     = (const int*)  d_in[0];
  const float* C     = (const float*)d_in[1];
  const float* W_ax  = (const float*)d_in[2];
  const float* W_aa  = (const float*)d_in[3];
  const float* W_aal = (const float*)d_in[4];
  const float* b_a   = (const float*)d_in[5];
  const float* W_out = (const float*)d_in[6];
  const float* b_out = (const float*)d_in[7];
  float* out = (float*)d_out;

  // Workspace layout (~34.3 MB)
  uint8_t* w = (uint8_t*)d_ws;
  size_t o = 0;
  __bf16* emb  = (__bf16*)(w + o); o += (size_t)SEQ * BATCH * EMB * sizeof(__bf16);
  __bf16* h0_0 = (__bf16*)(w + o); o += (size_t)BATCH * HID * sizeof(__bf16);
  __bf16* h0_1 = (__bf16*)(w + o); o += (size_t)BATCH * HID * sizeof(__bf16);
  __bf16* h1_0 = (__bf16*)(w + o); o += (size_t)BATCH * HID * sizeof(__bf16);
  __bf16* h1_1 = (__bf16*)(w + o); o += (size_t)BATCH * HID * sizeof(__bf16);
  float*  h1f  = (float*) (w + o); o += (size_t)BATCH * HID * sizeof(float);
  unsigned* ctrs = (unsigned*)(w + o); o += 256;

  // 1) Embedding gather into time-major bf16
  gather_emb_kernel<<<BATCH * SEQ, EMB, 0, stream>>>(X, C, emb);

  // 2) Zero h-state ping-pong buffers (contiguous from h0_0) + barrier ctrs
  const int nwords = (4 * BATCH * HID * (int)sizeof(__bf16)) / 4;  // 131072
  init_ws_kernel<<<(nwords + 255) / 256, 256, 0, stream>>>(ctrs, (unsigned*)h0_0, nwords);

  // 3) Persistent two-layer RNN scan (one grid barrier per timestep)
  rnn_persistent_kernel<<<NWG, 256, 0, stream>>>(W_ax, W_aa, W_aal, b_a, emb,
                                                 h0_0, h0_1, h1_0, h1_1, h1f, ctrs);

  // 4) fp32 WMMA projection to 32000 logits
  proj_kernel<<<N_CLASS / 16, 256, 0, stream>>>(h1f, W_out, b_out, out);
}
